// LSTMAdvancedAttentionContextOnly_26173530702471
// MI455X (gfx1250) — matrix operand order repaired and checked by reading, not compiled, not run on hardware
//
#include <hip/hip_runtime.h>
#include <hip/hip_bf16.h>
#include <math.h>

// ---------------------------------------------------------------------------
// LSTM + advanced attention, restructured for MI455X (gfx1250):
//  * x@W1x^T hoisted out of the 128-step recurrence (69 GFLOP -> 1.7 GFLOP)
//  * all GEMMs on v_wmma_f32_16x16x32_bf16 (wave32 WMMA, f32 accumulate)
//  * 8 persistent workgroups (one WGP each), weights sharded per WGP,
//    global monotonic atomic barrier (3/step) for the serial chain
// ---------------------------------------------------------------------------

typedef __bf16 v16bf __attribute__((ext_vector_type(16)));
typedef float  v8f   __attribute__((ext_vector_type(8)));

#define B_   8
#define T_   128
#define E_   256
#define H_   256
#define D_   512      // E_ + H_
#define NBLK 8
#define PTHREADS 512  // 16 waves per WGP

// ---- WMMA fragment packing per CDNA5 ISA 7.12.2 (16-bit A 16x32 / B 32x16) ----
template <typename F>
__device__ inline v16bf pack_a(int lane, int k0, F elem) {  // elem(row, k) -> float
  const int row  = lane & 15;
  const int half = lane >> 4;
  v16bf out;
#pragma unroll
  for (int v = 0; v < 8; ++v) {
    int kb = ((v < 4) ? 0 : 16) + (half ? 8 : 0) + ((v & 3) << 1);
    out[2 * v]     = (__bf16)elem(row, k0 + kb);
    out[2 * v + 1] = (__bf16)elem(row, k0 + kb + 1);
  }
  return out;
}
template <typename F>
__device__ inline v16bf pack_b(int lane, int k0, F elem) {  // elem(kk, n) -> float
  const int n    = lane & 15;
  const int half = lane >> 4;
  v16bf out;
#pragma unroll
  for (int v = 0; v < 8; ++v) {
    int kb = ((v < 4) ? 0 : 16) + (half ? 8 : 0) + ((v & 3) << 1);
    out[2 * v]     = (__bf16)elem(k0 + kb,     n);
    out[2 * v + 1] = (__bf16)elem(k0 + kb + 1, n);
  }
  return out;
}

__device__ inline float gload(float* p) {
  return __hip_atomic_load(p, __ATOMIC_RELAXED, __HIP_MEMORY_SCOPE_AGENT);
}
__device__ inline void gstore(float* p, float v) {
  __hip_atomic_store(p, v, __ATOMIC_RELAXED, __HIP_MEMORY_SCOPE_AGENT);
}
__device__ inline float sigm(float x) { return 1.0f / (1.0f + expf(-x)); }

// Monotonic-counter grid barrier (8 resident blocks; deterministic: counter is
// memset to 0 by the host each launch, target grows by NBLK per barrier).
__device__ inline void grid_barrier(unsigned* ctr, unsigned target) {
  __syncthreads();
  if (threadIdx.x == 0) {
    __threadfence();
    __hip_atomic_fetch_add(ctr, 1u, __ATOMIC_ACQ_REL, __HIP_MEMORY_SCOPE_AGENT);
    while (__hip_atomic_load(ctr, __ATOMIC_ACQUIRE, __HIP_MEMORY_SCOPE_AGENT) < target)
      __builtin_amdgcn_s_sleep(1);
  }
  __syncthreads();
  __threadfence();
}

// ---- kernel 0: f32 -> bf16 weight shards ----------------------------------
__global__ void convert_weights(const float* __restrict__ W1,
                                const float* __restrict__ Wih,
                                const float* __restrict__ Whh,
                                __bf16* W1hbf, __bf16* Wihbf, __bf16* Whhbf) {
  const int i = blockIdx.x * blockDim.x + threadIdx.x;
  if (i < D_ * H_) W1hbf[i] = (__bf16)W1[(i >> 8) * D_ + E_ + (i & 255)];  // W1[:, E:]
  if (i < 4 * H_ * E_) {
    Wihbf[i] = (__bf16)Wih[i];
    Whhbf[i] = (__bf16)Whh[i];
  }
}

// ---- kernel 1: xW1 = x @ W1[:, :E]^T + b1  (M=1024, N=512, K=256, WMMA) ----
__global__ __launch_bounds__(256)
void xw1_gemm(const float* __restrict__ x, const float* __restrict__ W1,
              const float* __restrict__ b1, __bf16* __restrict__ xW1bf) {
  const int lane = threadIdx.x & 31;
  const int wv   = blockIdx.x * (blockDim.x >> 5) + (threadIdx.x >> 5);  // 0..2047
  const int m0 = (wv >> 5) * 16;  // 64 row tiles
  const int n0 = (wv & 31) * 16;  // 32 col tiles
  v8f acc = {0.f, 0.f, 0.f, 0.f, 0.f, 0.f, 0.f, 0.f};
#pragma unroll
  for (int k0 = 0; k0 < E_; k0 += 32) {
    v16bf a  = pack_a(lane, k0, [&](int r, int k) -> float {
      return x[(m0 + r) * E_ + k]; });
    v16bf bm = pack_b(lane, k0, [&](int kk, int n) -> float {
      return W1[(n0 + n) * D_ + kk]; });
    acc = __builtin_amdgcn_wmma_f32_16x16x32_bf16(false, a, false, bm,
                                                  (short)0, acc, false, false);
  }
  const int   col  = n0 + (lane & 15);
  const float bias = b1[col];
  const int   mb   = m0 + ((lane >> 4) ? 8 : 0);
#pragma unroll
  for (int r = 0; r < 8; ++r)
    xW1bf[(mb + r) * D_ + col] = (__bf16)(acc[r] + bias);
}

// ---- kernel 2: persistent recurrence over t = 0..127 ----------------------
__global__ __launch_bounds__(PTHREADS)
void attn_lstm_persistent(
    const float* __restrict__ x, const int* __restrict__ seqlen,
    const float* __restrict__ W2, const float* __restrict__ b2,
    const float* __restrict__ bih, const float* __restrict__ bhh,
    const __bf16* __restrict__ xW1bf, const __bf16* __restrict__ W1hbf,
    const __bf16* __restrict__ Wihbf, const __bf16* __restrict__ Whhbf,
    float* hW_ws, float* ctx_ws, float* h_ws, unsigned* bar,
    float* out_outputs, float* out_seq, float* out_hidden, float* out_attn) {
  const int g    = blockIdx.x;       // owns batch row b=g (attention) and
  const int tid  = threadIdx.x;      // hidden-unit shard u in [32g, 32g+32)
  const int lane = tid & 31;
  const int wv   = tid >> 5;

  __shared__ float hl[B_ * H_];      // h_t (all batch rows)
  __shared__ float ctxall[B_ * E_];  // context (all batch rows)
  __shared__ float hWl[D_];          // hW[g, :]
  __shared__ float w2l[D_];
  __shared__ float el[T_];           // masked energies
  __shared__ float wl[T_];           // attention weights (exact zeros kept)
  __shared__ float ctxred[2 * E_];
  __shared__ float gatesl[4 * B_ * 32];
  __shared__ float cl[B_ * 32];      // cell-state shard (lives here all 128 steps)
  __shared__ float hlastl[B_ * 32];
  __shared__ int   llen[B_];

  if (tid < B_) llen[tid] = seqlen[tid];
  if (tid < D_) w2l[tid] = W2[tid];
  if (tid < B_ * 32) { cl[tid] = 0.f; hlastl[tid] = 0.f; }
  if (tid < B_ * 32) {  // zero this block's u-shard of the global hidden state
    const int b = tid >> 5, us = tid & 31;
    gstore(h_ws + b * H_ + g * 32 + us, 0.f);
  }
  const float b2v = b2[0];
  unsigned gen = 0;
  grid_barrier(bar, (++gen) * NBLK);

  for (int t = 0; t < T_; ++t) {
    // refresh h_t into LDS
    for (int i = tid; i < B_ * H_; i += PTHREADS) hl[i] = gload(h_ws + i);
    __syncthreads();

    // ---- phase 1: hW[:, 64g:64g+64] = h @ W1h^T shard (WMMA, waves 0..3) ----
    if (wv < 4) {
      const int ncol = g * 64 + wv * 16;
      v8f acc = {0.f, 0.f, 0.f, 0.f, 0.f, 0.f, 0.f, 0.f};
#pragma unroll
      for (int k0 = 0; k0 < H_; k0 += 32) {
        v16bf a  = pack_a(lane, k0, [&](int r, int k) -> float {
          return (r < B_) ? hl[r * H_ + k] : 0.0f; });
        v16bf bm = pack_b(lane, k0, [&](int kk, int n) -> float {
          return (float)W1hbf[(ncol + n) * H_ + kk]; });
        acc = __builtin_amdgcn_wmma_f32_16x16x32_bf16(false, a, false, bm,
                                                      (short)0, acc, false, false);
      }
      if (lane < 16) {
#pragma unroll
        for (int r = 0; r < B_; ++r) gstore(hW_ws + r * D_ + ncol + lane, acc[r]);
      }
    }
    grid_barrier(bar, (++gen) * NBLK);

    // ---- phase 2: energy -> softmax -> context for b = g -------------------
    for (int i = tid; i < D_; i += PTHREADS) hWl[i] = gload(hW_ws + g * D_ + i);
    __syncthreads();
    {
      const int row = tid >> 2, sub = tid & 3;  // 128 rows x 4 lanes
      const __bf16* xr = xW1bf + (g * T_ + row) * D_;
      float acc = 0.f;
      for (int i = 0; i < D_ / 4; ++i) {
        const int k = sub + i * 4;
        acc += tanhf((float)xr[k] + hWl[k]) * w2l[k];
      }
      acc += __shfl_xor(acc, 1, 4);
      acc += __shfl_xor(acc, 2, 4);
      if (sub == 0) el[row] = (row < llen[g]) ? tanhf(acc + b2v) : -1e30f;
    }
    __syncthreads();
    if (wv == 0) {  // masked softmax over 128 positions in one wave
      float e0 = el[lane], e1 = el[lane + 32], e2 = el[lane + 64], e3 = el[lane + 96];
      float m = fmaxf(fmaxf(e0, e1), fmaxf(e2, e3));
#pragma unroll
      for (int s = 16; s >= 1; s >>= 1) m = fmaxf(m, __shfl_xor(m, s, 32));
      float p0 = expf(e0 - m), p1 = expf(e1 - m), p2 = expf(e2 - m), p3 = expf(e3 - m);
      float s = p0 + p1 + p2 + p3;
#pragma unroll
      for (int d = 16; d >= 1; d >>= 1) s += __shfl_xor(s, d, 32);
      const int  L      = llen[g];
      const bool active = (t < L);
      const float inv   = active ? 1.0f / s : 0.0f;
      wl[lane]      = (active && lane      < L) ? p0 * inv : 0.f;
      wl[lane + 32] = (active && lane + 32 < L) ? p1 * inv : 0.f;
      wl[lane + 64] = (active && lane + 64 < L) ? p2 * inv : 0.f;
      wl[lane + 96] = (active && lane + 96 < L) ? p3 * inv : 0.f;
    }
    __syncthreads();
    if (tid < T_) out_attn[(g * T_ + t) * T_ + tid] = wl[tid];  // A[g, t, :]
    {
      const int e = tid & (E_ - 1), q = tid >> 8;
      float ca = 0.f;
      for (int r2 = q; r2 < T_; r2 += 2) ca += wl[r2] * x[(g * T_ + r2) * E_ + e];
      ctxred[q * E_ + e] = ca;
    }
    __syncthreads();
    if (tid < E_) gstore(ctx_ws + g * E_ + tid, ctxred[tid] + ctxred[E_ + tid]);
    grid_barrier(bar, (++gen) * NBLK);

    // ---- phase 3: gates = [ctx|h] @ [Wih|Whh]^T shard (WMMA) + LSTM update --
    for (int i = tid; i < B_ * E_; i += PTHREADS) ctxall[i] = gload(ctx_ws + i);
    __syncthreads();
    if (wv < 8) {
      const int q     = wv >> 1;                       // gate i/f/g/o
      const int basej = q * 256 + g * 32 + (wv & 1) * 16;
      v8f acc = {0.f, 0.f, 0.f, 0.f, 0.f, 0.f, 0.f, 0.f};
#pragma unroll
      for (int k0 = 0; k0 < 2 * H_; k0 += 32) {
        v16bf a = pack_a(lane, k0, [&](int r, int k) -> float {
          if (r >= B_) return 0.0f;
          return (k < E_) ? ctxall[r * E_ + k] : hl[r * H_ + (k - E_)]; });
        const __bf16* Bp  = (k0 < E_) ? Wihbf : Whhbf;
        const int kbase   = k0 & (E_ - 1);
        v16bf bm = pack_b(lane, k0, [&](int kk, int n) -> float {
          return (float)Bp[(basej + n) * E_ + kbase + (kk - k0)]; });
        acc = __builtin_amdgcn_wmma_f32_16x16x32_bf16(false, a, false, bm,
                                                      (short)0, acc, false, false);
      }
      if (lane < 16) {
        const int   j    = basej + lane;
        const float bias = bih[j] + bhh[j];
        const int   us   = (wv & 1) * 16 + lane;
#pragma unroll
        for (int r = 0; r < B_; ++r)
          gatesl[(q * B_ + r) * 32 + us] = acc[r] + bias;
      }
    }
    __syncthreads();
    if (tid < B_ * 32) {
      const int b = tid >> 5, us = tid & 31, u = g * 32 + us;
      const float gi = gatesl[(0 * B_ + b) * 32 + us];
      const float gf = gatesl[(1 * B_ + b) * 32 + us];
      const float gg = gatesl[(2 * B_ + b) * 32 + us];
      const float go = gatesl[(3 * B_ + b) * 32 + us];
      const float cn = sigm(gf) * cl[tid] + sigm(gi) * tanhf(gg);
      cl[tid] = cn;
      const float hn = sigm(go) * tanhf(cn);
      gstore(h_ws + b * H_ + u, hn);
      const int L = llen[b];
      out_outputs[(b * T_ + t) * H_ + u] = (t < L) ? hn : 0.0f;  // valid mask
      if (t == L - 1) hlastl[tid] = hn;
    }
    grid_barrier(bar, (++gen) * NBLK);
  }

  if (tid < B_ * 32) {
    const int b = tid >> 5, us = tid & 31;
    out_hidden[b * H_ + g * 32 + us] = hlastl[tid];  // hidden_state[0, b, :]
  }
  if (g == 0 && tid < B_) out_seq[tid] = (float)llen[tid];
}

// ---------------------------------------------------------------------------
extern "C" void kernel_launch(void* const* d_in, const int* in_sizes, int n_in,
                              void* d_out, int out_size, void* d_ws, size_t ws_size,
                              hipStream_t stream) {
  (void)in_sizes; (void)n_in; (void)out_size; (void)ws_size;
  const float* x      = (const float*)d_in[0];
  const int*   seqlen = (const int*)  d_in[1];
  const float* W1     = (const float*)d_in[2];
  const float* b1     = (const float*)d_in[3];
  const float* W2     = (const float*)d_in[4];
  const float* b2     = (const float*)d_in[5];
  const float* Wih    = (const float*)d_in[6];
  const float* Whh    = (const float*)d_in[7];
  const float* bih    = (const float*)d_in[8];
  const float* bhh    = (const float*)d_in[9];

  // workspace layout (~2.4 MB total)
  char* p = (char*)d_ws;
  __bf16* xW1bf = (__bf16*)p;  p += (size_t)(B_ * T_) * D_ * 2;  // 1 MB
  __bf16* W1hbf = (__bf16*)p;  p += (size_t)D_ * H_ * 2;         // 256 KB
  __bf16* Wihbf = (__bf16*)p;  p += (size_t)4 * H_ * E_ * 2;     // 512 KB
  __bf16* Whhbf = (__bf16*)p;  p += (size_t)4 * H_ * H_ * 2;     // 512 KB
  float* hW_ws  = (float*)p;   p += (size_t)B_ * D_ * 4;
  float* ctx_ws = (float*)p;   p += (size_t)B_ * E_ * 4;
  float* h_ws   = (float*)p;   p += (size_t)B_ * H_ * 4;
  unsigned* bar = (unsigned*)p;

  float* outp  = (float*)d_out;            // outputs [B,T,H]
  float* seqp  = outp + B_ * T_ * H_;      // seq_lengths [B] (as float)
  float* hidp  = seqp + B_;                // hidden_state [1,B,H]
  float* attnp = hidp + B_ * H_;           // attn [B,T,T]

  hipMemsetAsync(bar, 0, 256, stream);     // reset grid barrier each launch
  convert_weights<<<1024, 256, 0, stream>>>(W1, Wih, Whh, W1hbf, Wihbf, Whhbf);
  xw1_gemm<<<256, 256, 0, stream>>>(x, W1, b1, xW1bf);
  attn_lstm_persistent<<<NBLK, PTHREADS, 0, stream>>>(
      x, seqlen, W2, b2, bih, bhh, xW1bf, W1hbf, Wihbf, Whhbf,
      hW_ws, ctx_ws, h_ws, bar, outp, seqp, hidp, attnp);
}
